// IntraGraphAttention_11647951307189
// MI455X (gfx1250) — matrix operand order repaired and checked by reading, not compiled, not run on hardware
//
#include <hip/hip_runtime.h>
#include <hip/hip_bf16.h>

// ---------------------------------------------------------------------------
// GAT layer (IntraGraphAttention) for MI455X / gfx1250 (wave32, WMMA)
//
// Pipeline:
//  K0 gat_wswizzle  : W -> f16 fragment buffer in B-matrix register layout
//  K1 gat_proj_wmma : h = elu(x) @ W          -- v_wmma_f32_16x16x32_f16
//  K2 gat_logits    : a_s, a_d per node; init segment-max with self-loop logit
//  K3 gat_edge_max  : atomic-max of leaky_relu(a_s[src]+a_d[dst]) per dst
//  K4 gat_self_init : denom = exp_self; d_out = exp_self * h[n]
//  K5 gat_edge_agg  : one wave per edge; d_out[dst] += e_exp * h[src]
//  K6 gat_finalize  : d_out = d_out / denom + bias
// ---------------------------------------------------------------------------

#define N_NODES   100000
#define N_EDGES   1600000
#define IN_DIM    128
#define HEADS     2
#define OUT_CH    35
#define OUTC      (HEADS * OUT_CH)   // 70
#define NPAD      80                 // 5 tiles of 16 columns
#define NTILES    (N_NODES / 16)     // 6250 (exact)
#define NEG_SLOPE 0.2f
#define WF_HALFS  (5 * 4 * 32 * 16)  // 10240 f16 = 20 KB fragment buffer

typedef __attribute__((ext_vector_type(16))) _Float16 v16h;
typedef __attribute__((ext_vector_type(8)))  float    v8f;

// monotone float -> uint transform (order preserving) for atomic max
__device__ __forceinline__ unsigned int ford(float f) {
    unsigned int u = __float_as_uint(f);
    return (u & 0x80000000u) ? ~u : (u | 0x80000000u);
}
__device__ __forceinline__ float funord(unsigned int u) {
    return __uint_as_float((u & 0x80000000u) ? (u & 0x7fffffffu) : ~u);
}
__device__ __forceinline__ float elu1(float v) {
    return v > 0.f ? v : __expf(v) - 1.f;
}
__device__ __forceinline__ float lrelu(float v) {
    return v > 0.f ? v : NEG_SLOPE * v;
}

// ---------------------------------------------------------------------------
// K0: swizzle W (128x70 fp32) into per-fragment f16 layout:
//     wf[(nt*4+kf)*32 + lane][elem], 16 halfs (32 B) contiguous per lane.
// B-matrix layout (32x16 f16): lane<16: col=lane,    elem i -> K = kf*32+i
//                              lane>=16: col=lane-16, elem i -> K = kf*32+16+i
// ---------------------------------------------------------------------------
__global__ void gat_wswizzle(const float* __restrict__ W, _Float16* __restrict__ wf)
{
    const int idx = blockIdx.x * blockDim.x + threadIdx.x;   // over 128*80
    if (idx >= IN_DIM * NPAD) return;
    const int k = idx / NPAD, n = idx - k * NPAD;
    const float v = (n < OUTC) ? W[k * OUTC + n] : 0.f;
    const int nt = n >> 4, col = n & 15;
    const int kf = k >> 5, kk = k & 31;
    const int lane = col + (kk & 16);        // kk>=16 -> upper half-wave
    const int elem = kk & 15;
    wf[(size_t)((nt * 4 + kf) * 32 + lane) * 16 + elem] = (_Float16)v;
}

// ---------------------------------------------------------------------------
// K1: projection GEMM with WMMA. One wave per 16-node tile, 8 waves / block.
// A (16x32 f16) : lane<16: row=lane,     elems 0..15 -> K=k0+0..15
//                 lane>=16: row=lane-16, elems 0..15 -> K=k0+16..31
// C (16x16 f32) : vgpr r, lane<16 -> M=r, lane>=16 -> M=r+8, N=lane&15
// B fragments arrive pre-swizzled: one aligned 32 B vector load per fragment.
// ---------------------------------------------------------------------------
__global__ __launch_bounds__(256) void gat_proj_wmma(
        const float* __restrict__ x, const _Float16* __restrict__ wf,
        float* __restrict__ h)
{
    const int lane = threadIdx.x & 31;
    const int tile = blockIdx.x * 8 + (threadIdx.x >> 5);   // wave-uniform
    if (tile >= NTILES) return;                              // whole wave exits

    const int m0      = tile * 16;
    const int hi_half = (lane >= 16) ? 1 : 0;
    const int row     = lane & 15;
    const float4* xr4 = (const float4*)(x + (size_t)(m0 + row) * IN_DIM);
    const v16h*   wfp = (const v16h*)wf;     // 32 B fragments, 32 B aligned

    // A fragments for the whole K=128 (elu applied on the fly)
    v16h a[4];
#pragma unroll
    for (int kf = 0; kf < 4; ++kf) {
        const int k0 = kf * 32;
        const int b1 = (k0 + (hi_half ? 16 : 0)) >> 2;       // float4 index
        const int b2 = (k0 + 8 + (hi_half ? 16 : 0)) >> 2;
        const float4 r1a = xr4[b1],     r1b = xr4[b1 + 1];
        const float4 r2a = xr4[b2],     r2b = xr4[b2 + 1];
        const float r1[8] = {r1a.x, r1a.y, r1a.z, r1a.w, r1b.x, r1b.y, r1b.z, r1b.w};
        const float r2[8] = {r2a.x, r2a.y, r2a.z, r2a.w, r2b.x, r2b.y, r2b.z, r2b.w};
#pragma unroll
        for (int i = 0; i < 8; ++i) {
            a[kf][i]     = (_Float16)elu1(r1[i]);
            a[kf][i + 8] = (_Float16)elu1(r2[i]);
        }
    }

#pragma unroll
    for (int nt = 0; nt < 5; ++nt) {
        v16h b[4];
#pragma unroll
        for (int kf = 0; kf < 4; ++kf)
            b[kf] = wfp[(nt * 4 + kf) * 32 + lane];          // 2x global_load_b128
        v8f c = {};
#pragma unroll
        for (int kf = 0; kf < 4; ++kf)
            c = __builtin_amdgcn_wmma_f32_16x16x32_f16(
                    false, a[kf], false, b[kf], (short)0, c, false, false);
        const int n = nt * 16 + row;
        if (n < OUTC) {
#pragma unroll
            for (int r = 0; r < 8; ++r) {
                const int m = m0 + (hi_half ? r + 8 : r);
                h[(size_t)m * OUTC + n] = c[r];
            }
        }
    }
}

// ---------------------------------------------------------------------------
// K2: per-node attention logits + init segment max with the self-loop logit
// ---------------------------------------------------------------------------
__global__ void gat_logits(const float* __restrict__ h,
                           const float* __restrict__ a_src,
                           const float* __restrict__ a_dst,
                           float* __restrict__ as_buf, float* __restrict__ ad_buf,
                           unsigned int* __restrict__ emax)
{
    const int n = blockIdx.x * blockDim.x + threadIdx.x;
    if (n >= N_NODES) return;
    const float* hr = h + (size_t)n * OUTC;
#pragma unroll
    for (int hd = 0; hd < HEADS; ++hd) {
        float s = 0.f, d = 0.f;
#pragma unroll 5
        for (int c = 0; c < OUT_CH; ++c) {
            const float v = hr[hd * OUT_CH + c];
            s += v * a_src[hd * OUT_CH + c];
            d += v * a_dst[hd * OUT_CH + c];
        }
        as_buf[n * 2 + hd] = s;
        ad_buf[n * 2 + hd] = d;
        emax[n * 2 + hd]   = ford(lrelu(s + d));   // self loop seeds the max
    }
}

// ---------------------------------------------------------------------------
// K3: segment max over edges (atomic max on order-preserving uint)
// ---------------------------------------------------------------------------
__global__ void gat_edge_max(const int* __restrict__ ei,
                             const float* __restrict__ as_buf,
                             const float* __restrict__ ad_buf,
                             unsigned int* __restrict__ emax)
{
    const int e = blockIdx.x * blockDim.x + threadIdx.x;
    if (e >= N_EDGES) return;
    const int src = ei[e];
    const int dst = ei[N_EDGES + e];
#pragma unroll
    for (int hd = 0; hd < HEADS; ++hd) {
        const float v = lrelu(as_buf[src * 2 + hd] + ad_buf[dst * 2 + hd]);
        atomicMax(&emax[dst * 2 + hd], ford(v));
    }
}

// ---------------------------------------------------------------------------
// K4: self-loop contribution: denom = exp_self, out = exp_self * h[n]
// (out holds the UNNORMALIZED weighted sum; normalized in K6)
// ---------------------------------------------------------------------------
__global__ void gat_self_init(const float* __restrict__ h,
                              const float* __restrict__ as_buf,
                              const float* __restrict__ ad_buf,
                              const unsigned int* __restrict__ emax,
                              float* __restrict__ denom, float* __restrict__ out)
{
    const int n = blockIdx.x * blockDim.x + threadIdx.x;
    if (n >= N_NODES) return;
    float w[HEADS];
#pragma unroll
    for (int hd = 0; hd < HEADS; ++hd) {
        const float e  = lrelu(as_buf[n * 2 + hd] + ad_buf[n * 2 + hd]);
        w[hd]          = __expf(e - funord(emax[n * 2 + hd]));
        denom[n * 2 + hd] = w[hd];
    }
    const float* hr = h + (size_t)n * OUTC;
    float* outr     = out + (size_t)n * OUTC;
#pragma unroll 10
    for (int j = 0; j < OUTC; ++j)
        outr[j] = w[j / OUT_CH] * hr[j];
}

// ---------------------------------------------------------------------------
// K5: edge aggregation — one wave32 per edge, lane = output channel.
// h + out accumulator fit in the 192 MB L2, so the atomics stay on-chip.
// ---------------------------------------------------------------------------
__global__ __launch_bounds__(256) void gat_edge_agg(
        const int* __restrict__ ei,
        const float* __restrict__ h,
        const float* __restrict__ as_buf, const float* __restrict__ ad_buf,
        const unsigned int* __restrict__ emax,
        float* __restrict__ denom, float* __restrict__ out)
{
    const int lane = threadIdx.x & 31;
    const long long e = (long long)blockIdx.x * 8 + (threadIdx.x >> 5);
    if (e >= N_EDGES) return;
    const int src = ei[e];
    const int dst = ei[N_EDGES + e];

    const float* hr = h + (size_t)src * OUTC;
    __builtin_prefetch(hr, 0, 0);                   // global_prefetch_b8

    float w[HEADS];
#pragma unroll
    for (int hd = 0; hd < HEADS; ++hd) {
        const float v = lrelu(as_buf[src * 2 + hd] + ad_buf[dst * 2 + hd]);
        w[hd] = __expf(v - funord(emax[dst * 2 + hd]));
    }
    if (lane < HEADS) atomicAdd(&denom[dst * 2 + lane], w[lane]);

    float* outr = out + (size_t)dst * OUTC;
#pragma unroll
    for (int j = lane; j < OUTC; j += 32)
        atomicAdd(&outr[j], w[j / OUT_CH] * hr[j]);
}

// ---------------------------------------------------------------------------
// K6: normalize + bias
// ---------------------------------------------------------------------------
__global__ void gat_finalize(const float* __restrict__ denom,
                             const float* __restrict__ bias,
                             float* __restrict__ out)
{
    const long long idx = (long long)blockIdx.x * blockDim.x + threadIdx.x;
    if (idx >= (long long)N_NODES * OUTC) return;
    const int n = (int)(idx / OUTC);
    const int j = (int)(idx - (long long)n * OUTC);
    out[idx] = out[idx] / denom[n * 2 + (j >= OUT_CH ? 1 : 0)] + bias[j];
}

// ---------------------------------------------------------------------------
extern "C" void kernel_launch(void* const* d_in, const int* in_sizes, int n_in,
                              void* d_out, int out_size, void* d_ws, size_t ws_size,
                              hipStream_t stream)
{
    const float* x      = (const float*)d_in[0];
    const int*   ei     = (const int*)d_in[1];     // edge_index [2, E]
    const float* W      = (const float*)d_in[2];
    const float* a_src  = (const float*)d_in[3];
    const float* a_dst  = (const float*)d_in[4];
    const float* bias   = (const float*)d_in[5];
    float* out          = (float*)d_out;

    // workspace layout (wf first => 32 B alignment for v16h loads)
    _Float16* wf        = (_Float16*)d_ws;                      // 10,240 f16
    float* h            = (float*)d_ws + (WF_HALFS / 2);        // 7,000,000 f32
    float* as_buf       = h + (size_t)N_NODES * OUTC;           //   200,000
    float* ad_buf       = as_buf + (size_t)N_NODES * HEADS;     //   200,000
    unsigned int* emax  = (unsigned int*)(ad_buf + (size_t)N_NODES * HEADS);
    float* denom        = (float*)(emax + (size_t)N_NODES * HEADS);

    // K0: one-shot W swizzle into B-fragment layout (20 KB, L2-resident)
    gat_wswizzle<<<(IN_DIM * NPAD + 255) / 256, 256, 0, stream>>>(W, wf);

    // K1: WMMA projection (6250 tiles, 8 waves / block)
    gat_proj_wmma<<<(NTILES + 7) / 8, 256, 0, stream>>>(x, wf, h);

    // K2: logits + max init
    gat_logits<<<(N_NODES + 255) / 256, 256, 0, stream>>>(h, a_src, a_dst,
                                                          as_buf, ad_buf, emax);
    // K3: edge segment-max
    gat_edge_max<<<(N_EDGES + 255) / 256, 256, 0, stream>>>(ei, as_buf, ad_buf, emax);

    // K4: self-loop exp + accumulator init
    gat_self_init<<<(N_NODES + 255) / 256, 256, 0, stream>>>(h, as_buf, ad_buf,
                                                             emax, denom, out);
    // K5: edge aggregation (one wave per edge; 8 waves / 256-thread block)
    gat_edge_agg<<<(N_EDGES + 7) / 8, 256, 0, stream>>>(ei, h, as_buf, ad_buf,
                                                        emax, denom, out);
    // K6: normalize + bias
    const long long total = (long long)N_NODES * OUTC;
    gat_finalize<<<(int)((total + 255) / 256), 256, 0, stream>>>(denom, bias, out);
}
